// VisCorefAttender_27831388078163
// MI455X (gfx1250) — matrix operand order repaired and checked by reading, not compiled, not run on hardware
//
#include <hip/hip_runtime.h>
#include <hip/hip_bf16.h>

// ---------------------------------------------------------------------------
// VisCorefAttender on MI455X (gfx1250, wave32, WMMA + TDM).
// Dominant work: (B*M*C=73728) x 512 x 512 GEMM -> bf16 WMMA, f32 accumulate.
// ---------------------------------------------------------------------------

typedef __attribute__((ext_vector_type(16))) __bf16 v16bf;
typedef __attribute__((ext_vector_type(8)))  __bf16 v8bf;
typedef __attribute__((ext_vector_type(8)))  float  v8f;
typedef __attribute__((ext_vector_type(4)))  unsigned int u32x4;
typedef __attribute__((ext_vector_type(4)))  int i32x4;
typedef __attribute__((ext_vector_type(8)))  int i32x8;

static constexpr int Bb = 32, Mm = 64, Cc = 36;
static constexpr int IN_DIM = 1024, H = 512, NCLS = 1000, NCLS_PAD = 1024;
static constexpr float NEG_INF = -1e10f;

#if __has_builtin(__builtin_amdgcn_tensor_load_to_lds)
#define USE_TDM 1
#else
#define USE_TDM 0
#endif

// ---- WMMA wrapper: D = A(16x32 bf16) * B(32x16 bf16) + C(f32) ------------
__device__ __forceinline__ v8f wmma_bf16(v16bf a, v16bf b, v8f c) {
  return __builtin_amdgcn_wmma_f32_16x16x32_bf16(
      /*neg_a=*/false, a, /*neg_b=*/false, b,
      /*c_mod=*/(short)0, c, /*reuse_a=*/false, /*reuse_b=*/false);
}

// ---- A fragment: 16x32 bf16, row-major, stride lda elements --------------
// ISA layout (16-bit A 16x32): lane l holds row (l&15);
//   elems 0..7  = K = kb + 8*(l>>4) + 0..7
//   elems 8..15 = K = kb + 8*(l>>4) + 16..23
__device__ __forceinline__ v16bf load_frag_a(const __bf16* A, int lda,
                                             int row0, int kb, int lane) {
  const __bf16* p = A + (size_t)(row0 + (lane & 15)) * lda + kb + ((lane >> 4) << 3);
  v8bf lo = *(const v8bf*)(p);
  v8bf hi = *(const v8bf*)(p + 16);
  v16bf a;
#pragma unroll
  for (int i = 0; i < 8; ++i) { a[i] = lo[i]; a[i + 8] = hi[i]; }
  return a;
}

// ---- B fragment: 32x16 bf16 from N-major (transposed) weights ------------
// ISA layout (16-bit B 32x16): lane l holds column (l&15);
//   elems 0..7  = K = kb + 8*(l>>4) + 0..7
//   elems 8..15 = K = kb + 8*(l>>4) + 16..23
__device__ __forceinline__ v16bf load_frag_b(const __bf16* Wt, int ldk,
                                             int n0, int kb, int lane) {
  const __bf16* p = Wt + (size_t)(n0 + (lane & 15)) * ldk + kb + ((lane >> 4) << 3);
  v8bf lo = *(const v8bf*)(p);
  v8bf hi = *(const v8bf*)(p + 16);
  v16bf b;
#pragma unroll
  for (int i = 0; i < 8; ++i) { b[i] = lo[i]; b[i + 8] = hi[i]; }
  return b;
}

__device__ __forceinline__ v8f vzero8() {
  v8f z;
#pragma unroll
  for (int i = 0; i < 8; ++i) z[i] = 0.0f;
  return z;
}

// ---------------------------------------------------------------------------
// Prep kernels
// ---------------------------------------------------------------------------
__global__ void cvt_pad_kernel(const float* __restrict__ src, __bf16* __restrict__ dst,
                               int rows, int Ks, int Kd) {
  int idx = blockIdx.x * 256 + threadIdx.x;
  if (idx >= rows * Kd) return;
  int r = idx / Kd, k = idx - r * Kd;
  dst[idx] = (k < Ks) ? (__bf16)src[(size_t)r * Ks + k] : (__bf16)0.0f;
}

// w is (K x N) row-major; wt becomes (N x Kd) row-major, zero-padded in K.
__global__ void transpose_cvt_kernel(const float* __restrict__ w, __bf16* __restrict__ wt,
                                     int K, int N, int Kd) {
  int idx = blockIdx.x * 256 + threadIdx.x;
  if (idx >= N * Kd) return;
  int n = idx / Kd, k = idx - n * Kd;
  wt[idx] = (k < K) ? (__bf16)w[(size_t)k * N + n] : (__bf16)0.0f;
}

// ---------------------------------------------------------------------------
// Generic WMMA GEMM: D(rows x N) = act(A(rows x K) @ W + bias), bf16 out.
// Block = 256 threads = 8 waves; block tile 64 x 128.
// ---------------------------------------------------------------------------
template <bool RELU>
__global__ __launch_bounds__(256)
void gemm_bias_bf16(const __bf16* __restrict__ A, const __bf16* __restrict__ Wt,
                    const float* __restrict__ bias, __bf16* __restrict__ D,
                    int K, int N) {
  const int lane = threadIdx.x & 31;
  const int wv = threadIdx.x >> 5;
  const int row0 = blockIdx.x * 64 + (wv >> 1) * 16;
  const int col0 = blockIdx.y * 128 + (wv & 1) * 64;

  v8f acc[4];
#pragma unroll
  for (int t = 0; t < 4; ++t) acc[t] = vzero8();

  for (int kb = 0; kb < K; kb += 32) {
    v16bf a = load_frag_a(A, K, row0, kb, lane);
#pragma unroll
    for (int t = 0; t < 4; ++t) {
      v16bf b = load_frag_b(Wt, K, col0 + t * 16, kb, lane);
      acc[t] = wmma_bf16(a, b, acc[t]);
    }
  }

  const int rbase = row0 + ((lane >> 4) << 3);
#pragma unroll
  for (int t = 0; t < 4; ++t) {
    const int n = col0 + t * 16 + (lane & 15);
    const float bv = bias[n];
#pragma unroll
    for (int r = 0; r < 8; ++r) {
      float v = acc[t][r] + bv;
      if (RELU) v = fmaxf(v, 0.0f);
      D[(size_t)(rbase + r) * N + n] = (__bf16)v;
    }
  }
}

// ---------------------------------------------------------------------------
// Attention kernel: one block per (b, c).
//  - TDM (tensor_load_to_lds) DMAs the 64x512 bf16 men tile into LDS with
//    hardware row padding (pad_interval=7 -> every 256 DWORDs, pad_amount=3
//    -> 4 DWORDs) so rows land at stride 520 elems (conflict-free b128 ds).
//  - Threads form prod = men*vis in place (bf16), then each wave runs a
//    2 row-tile x 2 col-tile WMMA K-loop (2 A ds + 2 B global per 4 WMMAs),
//    fusing bias+ReLU+aw2-dot, lane-shuffle reduce, ds_add_f32 combine.
// ---------------------------------------------------------------------------
static constexpr int LDSK = H + 8;  // 520 elems / row, 1040 B

__global__ __launch_bounds__(256)
void attn_kernel(const __bf16* __restrict__ men, const __bf16* __restrict__ vis,
                 const __bf16* __restrict__ aw1t, const float* __restrict__ ab1,
                 const float* __restrict__ aw2, const float* __restrict__ ab2,
                 const float* __restrict__ mmask, const float* __restrict__ cmask,
                 float* __restrict__ out) {
  extern __shared__ char smem[];
  __bf16* prod = (__bf16*)smem;                               // 64 x LDSK bf16
  float* attn_s = (float*)(smem + (size_t)Mm * LDSK * 2);     // 64 f32

  const int bc = blockIdx.x;
  const int b = bc / Cc, c = bc - b * Cc;
  const int tid = threadIdx.x;

  const __bf16* menb = men + (size_t)b * Mm * H;
  const __bf16* visr = vis + (size_t)bc * H;

  if (tid < Mm) attn_s[tid] = 0.0f;

#if USE_TDM
  if (tid < 32) {  // wave 0 issues one TDM op for the whole block
    const unsigned lds_off = (unsigned)(uintptr_t)prod;          // LDS byte offset
    const unsigned long long ga = (unsigned long long)(uintptr_t)menb;
    u32x4 g0;
    g0[0] = 1u;                                                  // count=1 (valid)
    g0[1] = lds_off;                                             // lds_addr
    g0[2] = (unsigned)(ga & 0xffffffffu);                        // global_addr lo
    g0[3] = (unsigned)((ga >> 32) & 0x01ffffffu) | (2u << 30);   // ga hi | type=2
    i32x8 g1;
    g1[0] = (int)((1u << 16) |        // data_size = 2 bytes
                  (1u << 20) |        // pad_enable
                  (7u << 22) |        // pad_interval: 256 DWORDs (one 1024B row)
                  (3u << 25));        // pad_amount: 4 DWORDs (16 B)
    g1[1] = (int)((unsigned)(H & 0xffff) << 16);    // tensor_dim0 lo16 in [63:48]
    g1[2] = (int)(((unsigned)H >> 16) | ((unsigned)Mm << 16)); // dim0 hi | dim1 lo
    g1[3] = (int)(((unsigned)Mm >> 16) | ((unsigned)H << 16)); // dim1 hi | tile_dim0
    g1[4] = (int)(unsigned)Mm;                       // tile_dim1 (tile_dim2 = 0)
    g1[5] = (int)(unsigned)H;                        // tensor_dim0_stride lo32
    g1[6] = 0;
    g1[7] = 0;
    i32x4 gz = {0, 0, 0, 0};
#if defined(__clang_major__) && (__clang_major__ >= 23)
    i32x8 gz8 = {0, 0, 0, 0, 0, 0, 0, 0};
    __builtin_amdgcn_tensor_load_to_lds(g0, g1, gz, gz, gz8, 0);
#else
    __builtin_amdgcn_tensor_load_to_lds(g0, g1, gz, gz, 0);
#endif
    __builtin_amdgcn_s_wait_tensorcnt(0);
  }
  __syncthreads();
  // In-place prod = men * vis (bf16), men already staged by TDM.
  for (int idx = tid; idx < Mm * H; idx += 256) {
    const int r = idx >> 9, k = idx & (H - 1);
    const float p = (float)prod[r * LDSK + k] * (float)visr[k];
    prod[r * LDSK + k] = (__bf16)p;
  }
  __syncthreads();
#else
  for (int idx = tid; idx < Mm * H; idx += 256) {
    const int r = idx >> 9, k = idx & (H - 1);
    const float p = (float)menb[(size_t)r * H + k] * (float)visr[k];
    prod[r * LDSK + k] = (__bf16)p;
  }
  __syncthreads();
#endif

  const int lane = tid & 31;
  const int wv = tid >> 5;
  const int wr = wv >> 2;                 // row group: rows [32*wr, 32*wr+32)
  const int colq = wv & 3;                // column quarter: 128 cols

  float attn_acc[2][8];
#pragma unroll
  for (int s = 0; s < 2; ++s)
#pragma unroll
    for (int r = 0; r < 8; ++r) attn_acc[s][r] = 0.0f;

  for (int g = 0; g < 4; ++g) {           // 4 groups of 2 column tiles
    const int cb = colq * 128 + g * 32;
    if (g < 3) {  // prefetch next group's weight columns (global_prefetch_b8)
      __builtin_prefetch(aw1t + (size_t)(cb + 32 + (lane & 15)) * H, 0, 1);
      __builtin_prefetch(aw1t + (size_t)(cb + 48 + (lane & 15)) * H, 0, 1);
    }
    v8f acc[2][2];
#pragma unroll
    for (int s = 0; s < 2; ++s)
#pragma unroll
      for (int t = 0; t < 2; ++t) acc[s][t] = vzero8();

    for (int kb = 0; kb < H; kb += 32) {
      v16bf a0 = load_frag_a(prod, LDSK, wr * 32, kb, lane);        // ds_load
      v16bf a1 = load_frag_a(prod, LDSK, wr * 32 + 16, kb, lane);   // ds_load
      v16bf b0 = load_frag_b(aw1t, H, cb, kb, lane);                // global
      v16bf b1 = load_frag_b(aw1t, H, cb + 16, kb, lane);           // global
      acc[0][0] = wmma_bf16(a0, b0, acc[0][0]);
      acc[0][1] = wmma_bf16(a0, b1, acc[0][1]);
      acc[1][0] = wmma_bf16(a1, b0, acc[1][0]);
      acc[1][1] = wmma_bf16(a1, b1, acc[1][1]);
    }
    // Fused epilogue: bias + ReLU + dot with aw2 (lane owns one column n).
#pragma unroll
    for (int t = 0; t < 2; ++t) {
      const int n = cb + t * 16 + (lane & 15);
      const float b1v = ab1[n], w2 = aw2[n];
#pragma unroll
      for (int s = 0; s < 2; ++s)
#pragma unroll
        for (int r = 0; r < 8; ++r) {
          float hv = acc[s][t][r] + b1v;
          hv = fmaxf(hv, 0.0f);
          attn_acc[s][r] += hv * w2;
        }
    }
  }

  // Reduce over the 16 lanes of each half-wave (columns); rows stay per-lane.
#pragma unroll
  for (int m = 8; m >= 1; m >>= 1) {
#pragma unroll
    for (int s = 0; s < 2; ++s)
#pragma unroll
      for (int r = 0; r < 8; ++r)
        attn_acc[s][r] += __shfl_xor(attn_acc[s][r], m, 32);
  }
  if ((lane & 15) == 0) {
#pragma unroll
    for (int s = 0; s < 2; ++s) {
      const int rbase = wr * 32 + s * 16 + ((lane >> 4) << 3);
#pragma unroll
      for (int r = 0; r < 8; ++r)
        atomicAdd(&attn_s[rbase + r], attn_acc[s][r]);  // ds_add_f32
    }
  }
  __syncthreads();

  if (tid < Mm) {
    float v = attn_s[tid] + ab2[0];
    v = v * mmask[b * Mm + tid] * cmask[b * Cc + c];
    v = (v != 0.0f) ? v : NEG_INF;
    out[(size_t)(b * Mm + tid) * Cc + c] = v;
  }
}

// ---------------------------------------------------------------------------
// Host launcher
// ---------------------------------------------------------------------------
extern "C" void kernel_launch(void* const* d_in, const int* in_sizes, int n_in,
                              void* d_out, int out_size, void* d_ws, size_t ws_size,
                              hipStream_t stream) {
  (void)in_sizes; (void)n_in; (void)out_size; (void)ws_size;

  const float* mention = (const float*)d_in[0];   // (32,64,1024)
  const float* clabels = (const float*)d_in[1];   // (32,36,1000)
  const float* mmask   = (const float*)d_in[2];   // (32,64)
  const float* cmask   = (const float*)d_in[3];   // (32,36)
  const float* vw1 = (const float*)d_in[4];  const float* vb1 = (const float*)d_in[5];
  const float* vw2 = (const float*)d_in[6];  const float* vb2 = (const float*)d_in[7];
  const float* mw1 = (const float*)d_in[8];  const float* mb1 = (const float*)d_in[9];
  const float* mw2 = (const float*)d_in[10]; const float* mb2 = (const float*)d_in[11];
  const float* aw1 = (const float*)d_in[12]; const float* ab1 = (const float*)d_in[13];
  const float* aw2 = (const float*)d_in[14]; const float* ab2 = (const float*)d_in[15];
  float* out = (float*)d_out;

  char* ws = (char*)d_ws;
  size_t off = 0;
  auto carve = [&](size_t elems) -> __bf16* {
    __bf16* p = (__bf16*)(ws + off);
    off = (off + elems * sizeof(__bf16) + 255) & ~(size_t)255;
    return p;
  };
  __bf16* A_m    = carve((size_t)Bb * Mm * IN_DIM);      // 2048 x 1024
  __bf16* A_c    = carve((size_t)Bb * Cc * NCLS_PAD);    // 1152 x 1024 (padded)
  __bf16* mw1t   = carve((size_t)H * IN_DIM);            // 512 x 1024
  __bf16* mw2t   = carve((size_t)H * H);
  __bf16* vw1t   = carve((size_t)H * NCLS_PAD);          // 512 x 1024 (padded)
  __bf16* vw2t   = carve((size_t)H * H);
  __bf16* aw1t   = carve((size_t)H * H);
  __bf16* tmp_m1 = carve((size_t)Bb * Mm * H);           // 2048 x 512
  __bf16* tmp_v1 = carve((size_t)Bb * Cc * H);           // 1152 x 512
  __bf16* men_bf = carve((size_t)Bb * Mm * H);
  __bf16* vis_bf = carve((size_t)Bb * Cc * H);

  const int rows_m = Bb * Mm;   // 2048
  const int rows_c = Bb * Cc;   // 1152

  // --- prep: convert activations + transpose/convert weights --------------
  cvt_pad_kernel<<<(rows_m * IN_DIM + 255) / 256, 256, 0, stream>>>(
      mention, A_m, rows_m, IN_DIM, IN_DIM);
  cvt_pad_kernel<<<(rows_c * NCLS_PAD + 255) / 256, 256, 0, stream>>>(
      clabels, A_c, rows_c, NCLS, NCLS_PAD);
  transpose_cvt_kernel<<<(H * IN_DIM + 255) / 256, 256, 0, stream>>>(
      mw1, mw1t, IN_DIM, H, IN_DIM);
  transpose_cvt_kernel<<<(H * H + 255) / 256, 256, 0, stream>>>(
      mw2, mw2t, H, H, H);
  transpose_cvt_kernel<<<(H * NCLS_PAD + 255) / 256, 256, 0, stream>>>(
      vw1, vw1t, NCLS, H, NCLS_PAD);
  transpose_cvt_kernel<<<(H * H + 255) / 256, 256, 0, stream>>>(
      vw2, vw2t, H, H, H);
  transpose_cvt_kernel<<<(H * H + 255) / 256, 256, 0, stream>>>(
      aw1, aw1t, H, H, H);

  // --- embedders (WMMA GEMMs) ----------------------------------------------
  dim3 blk(256);
  gemm_bias_bf16<true><<<dim3(rows_m / 64, H / 128), blk, 0, stream>>>(
      A_m, mw1t, mb1, tmp_m1, IN_DIM, H);
  gemm_bias_bf16<false><<<dim3(rows_m / 64, H / 128), blk, 0, stream>>>(
      tmp_m1, mw2t, mb2, men_bf, H, H);
  gemm_bias_bf16<true><<<dim3(rows_c / 64, H / 128), blk, 0, stream>>>(
      A_c, vw1t, vb1, tmp_v1, NCLS_PAD, H);
  gemm_bias_bf16<false><<<dim3(rows_c / 64, H / 128), blk, 0, stream>>>(
      tmp_v1, vw2t, vb2, vis_bf, H, H);

  // --- fused attention (dominant GEMM + epilogue) --------------------------
  const size_t attn_lds = (size_t)Mm * LDSK * sizeof(__bf16) + Mm * sizeof(float);
  attn_kernel<<<dim3(Bb * Cc), blk, attn_lds, stream>>>(
      men_bf, vis_bf, aw1t, ab1, aw2, ab2, mmask, cmask, out);
}
